// ScoreNetwork_56831007261036
// MI455X (gfx1250) — matrix operand order repaired and checked by reading, not compiled, not run on hardware
//
#include <hip/hip_runtime.h>
#include <hip/hip_bf16.h>
#include <math.h>

typedef __attribute__((ext_vector_type(16))) __bf16 v16bf;
typedef __attribute__((ext_vector_type(8)))  float  v8f;

constexpr int kH  = 256;   // hidden width
constexpr int kNL = 4;     // residual layers
constexpr int kN  = 8192;  // points per batch
constexpr int kB  = 4;     // batch
constexpr int kTR = 32;    // rows (points) per workgroup tile
constexpr int kABP = kH + 8; // padded bf16 activation row (bank-conflict dodge, keeps 16B align)

// workspace layout (bytes)
constexpr size_t OFF_TPROJ   = 0;                                   // 4*256 f32
constexpr size_t OFF_NEAREST = 4096;                                // 4*8192*3 f32
constexpr size_t OFF_W1      = OFF_NEAREST + (size_t)kB*kN*3*4;     // 4*256*256 bf16
constexpr size_t OFF_W2      = OFF_W1 + (size_t)kNL*kH*kH*2;
constexpr size_t OFF_OW1     = OFF_W2 + (size_t)kNL*kH*kH*2;        // 256*128 bf16

static __device__ __forceinline__ unsigned short f32_to_bf16(float f) {
  union { float f; unsigned u; } v; v.f = f;
  unsigned r = v.u + 0x7FFFu + ((v.u >> 16) & 1u);
  return (unsigned short)(r >> 16);
}
static __device__ __forceinline__ float silu_f(float x) {
  return x / (1.0f + __expf(-x));
}

union FragBF { v16bf v; uint4 q[2]; };

// ---------------- kernel 1: time embedding (tiny, one block) ----------------
__global__ __launch_bounds__(256)
void time_embed_kernel(const int* __restrict__ timesteps,
                       const float* __restrict__ te_w1, const float* __restrict__ te_b1,
                       const float* __restrict__ te_w2, const float* __restrict__ te_b2,
                       const float* __restrict__ tp_w,  const float* __restrict__ tp_b,
                       float* __restrict__ tproj) {
  __shared__ float emb[kB][128];
  __shared__ float h1[kB][512];
  __shared__ float tm[kB][512];
  const int t = threadIdx.x;
  const float factor = -logf(10000.0f) / 63.0f;
  for (int i = t; i < kB * 128; i += 256) {
    int b = i >> 7, j = i & 127;
    float freq = __expf((float)(j & 63) * factor);
    float ang = (float)timesteps[b] * freq;
    emb[b][j] = (j < 64) ? __sinf(ang) : __cosf(ang);
  }
  __syncthreads();
  for (int i = t; i < kB * 512; i += 256) {
    int b = i >> 9, j = i & 511;
    float s = te_b1[j];
    for (int k = 0; k < 128; ++k) s = fmaf(emb[b][k], te_w1[k * 512 + j], s);
    h1[b][j] = silu_f(s);
  }
  __syncthreads();
  for (int i = t; i < kB * 512; i += 256) {
    int b = i >> 9, j = i & 511;
    float s = te_b2[j];
    for (int k = 0; k < 512; ++k) s = fmaf(h1[b][k], te_w2[k * 512 + j], s);
    tm[b][j] = s;
  }
  __syncthreads();
  for (int i = t; i < kB * kH; i += 256) {
    int b = i >> 8, j = i & 255;
    float s = tp_b[j];
    for (int k = 0; k < 512; ++k) s = fmaf(tm[b][k], tp_w[k * kH + j], s);
    tproj[b * kH + j] = s;
  }
}

// ---------------- kernel 2: brute-force nearest neighbor ----------------
// argmin_j |p-t_j|^2 == argmin_j (|t_j|^2 - 2 p.t_j); pack (x,y,z,|t|^2) as float4
__global__ __launch_bounds__(256)
void nn_kernel(const float* __restrict__ noisy, const float* __restrict__ target,
               float* __restrict__ nearest) {
  __shared__ float4 tt[256];
  const int b = blockIdx.y;
  const int i = blockIdx.x * 256 + threadIdx.x;
  const float* np = noisy + ((size_t)b * kN + i) * 3;
  const float mx = -2.0f * np[0], my = -2.0f * np[1], mz = -2.0f * np[2];
  float best = 3.4e38f;
  int bj = 0;
  const float* tgt = target + (size_t)b * kN * 3;
  for (int j0 = 0; j0 < kN; j0 += 256) {
    __syncthreads();
    {
      int j = j0 + threadIdx.x;
      float x = tgt[j * 3 + 0], y = tgt[j * 3 + 1], z = tgt[j * 3 + 2];
      tt[threadIdx.x] = make_float4(x, y, z, fmaf(x, x, fmaf(y, y, z * z)));
    }
    __syncthreads();
    #pragma unroll 8
    for (int jj = 0; jj < 256; ++jj) {
      float4 q = tt[jj];
      float s = fmaf(mx, q.x, fmaf(my, q.y, fmaf(mz, q.z, q.w)));
      if (s < best) { best = s; bj = j0 + jj; }
    }
  }
  const float* bp = tgt + (size_t)bj * 3;
  float* o = nearest + ((size_t)b * kN + i) * 3;
  o[0] = bp[0]; o[1] = bp[1]; o[2] = bp[2];
}

// ------------- kernel 3: fp32 -> bf16 weight swizzle into WMMA B-fragment order -------------
// B (KxN): lane l holds col = l&15; element e holds K = kt*32 + (e>>3)*16 + (l>>4)*8 + (e&7)
__global__ __launch_bounds__(256)
void swizzle_kernel(const float* __restrict__ src, unsigned short* __restrict__ dst,
                    int K, int N) {
  int t = blockIdx.x * 256 + threadIdx.x;
  if (t >= K * N) return;
  int e    = t & 15;
  int lane = (t >> 4) & 31;
  int tile = t >> 9;
  int NT = N >> 4;
  int nt = tile % NT;
  int kt = tile / NT;
  int k = kt * 32 + (e >> 3) * 16 + (lane >> 4) * 8 + (e & 7);
  int n = nt * 16 + (lane & 15);
  dst[t] = f32_to_bf16(src[(size_t)k * N + n]);
}

// ---------------- kernel 4: fused WMMA residual MLP + head ----------------
// 32 points/block; residual stream x lives in registers in WMMA C-fragment layout.
__global__ __launch_bounds__(256)
void mlp_kernel(const float* __restrict__ noisy, const float* __restrict__ nearest,
                const float* __restrict__ tproj,
                const float* __restrict__ in_w, const float* __restrict__ in_b,
                const unsigned short* __restrict__ wsw1, const float* __restrict__ ly_b1,
                const float* __restrict__ ly_g,  const float* __restrict__ ly_be,
                const unsigned short* __restrict__ wsw2, const float* __restrict__ ly_b2,
                const unsigned short* __restrict__ wsow1, const float* __restrict__ out_b1,
                const float* __restrict__ out_w2, const float* __restrict__ out_b2,
                float* __restrict__ out) {
  __shared__ float hf[kTR][kH];                                          // 32 KB
  __shared__ __attribute__((aligned(16))) unsigned short ab[kTR][kABP];  // 16.5 KB

  const int t     = threadIdx.x;
  const int lane  = t & 31;
  const int wave  = t >> 5;
  const int khalf = lane >> 4;
  const int arow  = lane & 15;       // A-matrix row within 16-row subtile
  const int ncol  = lane & 15;       // C/D column within 16-col subtile
  const int crow  = khalf * 8;       // C/D base row for this half-wave
  const int row0  = blockIdx.x * kTR;
  const int batch = row0 / kN;
  const int n0    = wave * 32;       // this wave's 32 output columns
  const int nt0   = wave * 2;        // first of its two 16-col tiles

  // ----- input projection: x0 = [noisy,nearest] @ in_w + in_b + tproj[batch] -----
  {
    const int col = t;
    float base = in_b[col] + tproj[batch * kH + col];
    for (int r = 0; r < kTR; ++r) {
      const float* np = noisy   + (size_t)(row0 + r) * 3;
      const float* nr = nearest + (size_t)(row0 + r) * 3;
      float s = base;
      s = fmaf(np[0], in_w[0 * kH + col], s);
      s = fmaf(np[1], in_w[1 * kH + col], s);
      s = fmaf(np[2], in_w[2 * kH + col], s);
      s = fmaf(nr[0], in_w[3 * kH + col], s);
      s = fmaf(nr[1], in_w[4 * kH + col], s);
      s = fmaf(nr[2], in_w[5 * kH + col], s);
      hf[r][col] = s;
      ab[r][col] = f32_to_bf16(s);
    }
  }
  __syncthreads();

  // pull residual stream into registers (C-fragment layout)
  v8f xr[2][2];
  #pragma unroll
  for (int rs = 0; rs < 2; ++rs)
    #pragma unroll
    for (int nt = 0; nt < 2; ++nt)
      #pragma unroll
      for (int v = 0; v < 8; ++v)
        xr[rs][nt][v] = hf[rs * 16 + crow + v][n0 + nt * 16 + ncol];
  __syncthreads();  // hf is now free scratch

  for (int li = 0; li < kNL; ++li) {
    // ---- GEMM1: h = x @ w1 + b1 ----
    const unsigned short* w1 = wsw1 + (size_t)li * kH * kH;
    v8f c[2][2];
    {
      float bv0 = ly_b1[li * kH + n0 + ncol];
      float bv1 = ly_b1[li * kH + n0 + 16 + ncol];
      #pragma unroll
      for (int v = 0; v < 8; ++v) {
        c[0][0][v] = bv0; c[1][0][v] = bv0;
        c[0][1][v] = bv1; c[1][1][v] = bv1;
      }
    }
    #pragma unroll
    for (int kt = 0; kt < 8; ++kt) {
      const int k0 = kt * 32;
      FragBF b0, b1;
      const uint4* p0 = (const uint4*)(w1 + ((((size_t)kt * 16 + nt0    ) * 32 + lane) << 4));
      const uint4* p1 = (const uint4*)(w1 + ((((size_t)kt * 16 + nt0 + 1) * 32 + lane) << 4));
      b0.q[0] = p0[0]; b0.q[1] = p0[1];
      b1.q[0] = p1[0]; b1.q[1] = p1[1];
      #pragma unroll
      for (int rs = 0; rs < 2; ++rs) {
        FragBF a;
        a.q[0] = *(const uint4*)&ab[rs * 16 + arow][k0 + khalf * 8];
        a.q[1] = *(const uint4*)&ab[rs * 16 + arow][k0 + 16 + khalf * 8];
        c[rs][0] = __builtin_amdgcn_wmma_f32_16x16x32_bf16(false, a.v, false, b0.v, (short)0, c[rs][0], false, false);
        c[rs][1] = __builtin_amdgcn_wmma_f32_16x16x32_bf16(false, a.v, false, b1.v, (short)0, c[rs][1], false, false);
      }
    }
    #pragma unroll
    for (int rs = 0; rs < 2; ++rs)
      #pragma unroll
      for (int nt = 0; nt < 2; ++nt)
        #pragma unroll
        for (int v = 0; v < 8; ++v)
          hf[rs * 16 + crow + v][n0 + nt * 16 + ncol] = c[rs][nt][v];
    __syncthreads();

    // ---- LayerNorm + SiLU -> ab (bf16) ----
    {
      const int row = t >> 3;    // 8 threads per row
      const int sub = t & 7;
      float s = 0.f, q = 0.f;
      for (int j = 0; j < 32; ++j) {
        float v = hf[row][sub * 32 + j];
        s += v; q += v * v;
      }
      for (int m = 1; m < 8; m <<= 1) {
        s += __shfl_xor(s, m, 32);
        q += __shfl_xor(q, m, 32);
      }
      float mean = s * (1.0f / kH);
      float var  = q * (1.0f / kH) - mean * mean;
      float inv  = rsqrtf(var + 1e-5f);
      for (int j = 0; j < 32; ++j) {
        int cc = sub * 32 + j;
        float hn = (hf[row][cc] - mean) * inv * ly_g[li * kH + cc] + ly_be[li * kH + cc];
        ab[row][cc] = f32_to_bf16(silu_f(hn));
      }
    }
    __syncthreads();

    // ---- GEMM2: x = x + silu(h) @ w2 + b2  (accumulate straight into residual regs) ----
    const unsigned short* w2 = wsw2 + (size_t)li * kH * kH;
    {
      float bv0 = ly_b2[li * kH + n0 + ncol];
      float bv1 = ly_b2[li * kH + n0 + 16 + ncol];
      #pragma unroll
      for (int v = 0; v < 8; ++v) {
        c[0][0][v] = xr[0][0][v] + bv0; c[1][0][v] = xr[1][0][v] + bv0;
        c[0][1][v] = xr[0][1][v] + bv1; c[1][1][v] = xr[1][1][v] + bv1;
      }
    }
    #pragma unroll
    for (int kt = 0; kt < 8; ++kt) {
      const int k0 = kt * 32;
      FragBF b0, b1;
      const uint4* p0 = (const uint4*)(w2 + ((((size_t)kt * 16 + nt0    ) * 32 + lane) << 4));
      const uint4* p1 = (const uint4*)(w2 + ((((size_t)kt * 16 + nt0 + 1) * 32 + lane) << 4));
      b0.q[0] = p0[0]; b0.q[1] = p0[1];
      b1.q[0] = p1[0]; b1.q[1] = p1[1];
      #pragma unroll
      for (int rs = 0; rs < 2; ++rs) {
        FragBF a;
        a.q[0] = *(const uint4*)&ab[rs * 16 + arow][k0 + khalf * 8];
        a.q[1] = *(const uint4*)&ab[rs * 16 + arow][k0 + 16 + khalf * 8];
        c[rs][0] = __builtin_amdgcn_wmma_f32_16x16x32_bf16(false, a.v, false, b0.v, (short)0, c[rs][0], false, false);
        c[rs][1] = __builtin_amdgcn_wmma_f32_16x16x32_bf16(false, a.v, false, b1.v, (short)0, c[rs][1], false, false);
      }
    }
    __syncthreads();   // all waves done reading ab before we overwrite it
    #pragma unroll
    for (int rs = 0; rs < 2; ++rs)
      #pragma unroll
      for (int nt = 0; nt < 2; ++nt) {
        xr[rs][nt] = c[rs][nt];
        #pragma unroll
        for (int v = 0; v < 8; ++v)
          ab[rs * 16 + crow + v][n0 + nt * 16 + ncol] = f32_to_bf16(xr[rs][nt][v]);
      }
    __syncthreads();
  }

  // ---- head GEMM: 256 -> 128, then SiLU -> hf ----
  {
    v8f c0, c1;
    float bv = out_b1[wave * 16 + ncol];
    #pragma unroll
    for (int v = 0; v < 8; ++v) { c0[v] = bv; c1[v] = bv; }
    #pragma unroll
    for (int kt = 0; kt < 8; ++kt) {
      const int k0 = kt * 32;
      FragBF b;
      const uint4* p = (const uint4*)(wsow1 + ((((size_t)kt * 8 + wave) * 32 + lane) << 4));
      b.q[0] = p[0]; b.q[1] = p[1];
      FragBF a0, a1;
      a0.q[0] = *(const uint4*)&ab[arow][k0 + khalf * 8];
      a0.q[1] = *(const uint4*)&ab[arow][k0 + 16 + khalf * 8];
      a1.q[0] = *(const uint4*)&ab[16 + arow][k0 + khalf * 8];
      a1.q[1] = *(const uint4*)&ab[16 + arow][k0 + 16 + khalf * 8];
      c0 = __builtin_amdgcn_wmma_f32_16x16x32_bf16(false, a0.v, false, b.v, (short)0, c0, false, false);
      c1 = __builtin_amdgcn_wmma_f32_16x16x32_bf16(false, a1.v, false, b.v, (short)0, c1, false, false);
    }
    #pragma unroll
    for (int v = 0; v < 8; ++v) {
      hf[crow + v][wave * 16 + ncol]      = silu_f(c0[v]);
      hf[16 + crow + v][wave * 16 + ncol] = silu_f(c1[v]);
    }
  }
  __syncthreads();

  // ---- final 128 -> 3 ----
  if (t < kTR * 3) {
    int r = t / 3, p = t % 3;
    float s = out_b2[p];
    for (int k = 0; k < 128; ++k)
      s = fmaf(hf[r][k], out_w2[k * 3 + p], s);
    out[(size_t)(row0 + r) * 3 + p] = s;
  }
}

extern "C" void kernel_launch(void* const* d_in, const int* in_sizes, int n_in,
                              void* d_out, int out_size, void* d_ws, size_t ws_size,
                              hipStream_t stream) {
  const float* noisy     = (const float*)d_in[0];
  const float* target    = (const float*)d_in[1];
  const int*   timesteps = (const int*)  d_in[2];
  const float* te_w1  = (const float*)d_in[3];
  const float* te_b1  = (const float*)d_in[4];
  const float* te_w2  = (const float*)d_in[5];
  const float* te_b2  = (const float*)d_in[6];
  const float* tp_w   = (const float*)d_in[7];
  const float* tp_b   = (const float*)d_in[8];
  const float* in_w   = (const float*)d_in[9];
  const float* in_b   = (const float*)d_in[10];
  const float* ly_w1  = (const float*)d_in[11];
  const float* ly_b1  = (const float*)d_in[12];
  const float* ly_g   = (const float*)d_in[13];
  const float* ly_be  = (const float*)d_in[14];
  const float* ly_w2  = (const float*)d_in[15];
  const float* ly_b2  = (const float*)d_in[16];
  const float* out_w1 = (const float*)d_in[17];
  const float* out_b1 = (const float*)d_in[18];
  const float* out_w2 = (const float*)d_in[19];
  const float* out_b2 = (const float*)d_in[20];

  char* ws = (char*)d_ws;
  float*          tproj   = (float*)(ws + OFF_TPROJ);
  float*          nearest = (float*)(ws + OFF_NEAREST);
  unsigned short* wsw1    = (unsigned short*)(ws + OFF_W1);
  unsigned short* wsw2    = (unsigned short*)(ws + OFF_W2);
  unsigned short* wsow1   = (unsigned short*)(ws + OFF_OW1);

  time_embed_kernel<<<1, 256, 0, stream>>>(timesteps, te_w1, te_b1, te_w2, te_b2,
                                           tp_w, tp_b, tproj);
  nn_kernel<<<dim3(kN / 256, kB), 256, 0, stream>>>(noisy, target, nearest);
  for (int li = 0; li < kNL; ++li) {
    swizzle_kernel<<<(kH * kH + 255) / 256, 256, 0, stream>>>(
        ly_w1 + (size_t)li * kH * kH, wsw1 + (size_t)li * kH * kH, kH, kH);
    swizzle_kernel<<<(kH * kH + 255) / 256, 256, 0, stream>>>(
        ly_w2 + (size_t)li * kH * kH, wsw2 + (size_t)li * kH * kH, kH, kH);
  }
  swizzle_kernel<<<(kH * 128 + 255) / 256, 256, 0, stream>>>(out_w1, wsow1, kH, 128);

  mlp_kernel<<<(kB * kN) / kTR, 256, 0, stream>>>(
      noisy, nearest, tproj, in_w, in_b,
      wsw1, ly_b1, ly_g, ly_be, wsw2, ly_b2,
      wsow1, out_b1, out_w2, out_b2, (float*)d_out);
}
